// SageLayer_37787122270589
// MI455X (gfx1250) — compile-verified, matrix operand order checked by reading, not run on hardware
//
#include <hip/hip_runtime.h>
#include <hip/hip_bf16.h>

typedef __attribute__((ext_vector_type(16))) __bf16 v16bf;
typedef __attribute__((ext_vector_type(8)))  __bf16 v8bf;
typedef __attribute__((ext_vector_type(8)))  float  v8f;

#define N_ROWS 8192
#define F_DIM  1024
#define O_DIM  1024
#define K2     2048   // 2*F
#define NNEIGH 10

union FragB16 { v16bf v; v8bf h[2]; };

// ---------------------------------------------------------------------------
// Kernel 1: combined[n, 0:F] = bf16(X[n, :]); combined[n, F:2F] = bf16(mean10)
// ---------------------------------------------------------------------------
__global__ void __launch_bounds__(256)
build_combined_kernel(const float* __restrict__ X,
                      const int* __restrict__ idx,
                      __bf16* __restrict__ combined) {
    const int t   = blockIdx.x * 256 + threadIdx.x;
    const int row = t >> 8;              // 256 threads cover one row (2048/8)
    const int seg = (t & 255) * 8;       // column within [0, 2F)
    __bf16* dst = combined + (size_t)row * K2 + seg;

    float acc[8];
    if (seg < F_DIM) {
        const float* src = X + (size_t)row * F_DIM + seg;
        const float4 a = *(const float4*)(src);
        const float4 b = *(const float4*)(src + 4);
        acc[0]=a.x; acc[1]=a.y; acc[2]=a.z; acc[3]=a.w;
        acc[4]=b.x; acc[5]=b.y; acc[6]=b.z; acc[7]=b.w;
    } else {
        const int f = seg - F_DIM;
        #pragma unroll
        for (int i = 0; i < 8; ++i) acc[i] = 0.0f;
        #pragma unroll
        for (int j = 0; j < NNEIGH; ++j) {
            const int nb = idx[row * NNEIGH + j];
            const float* src = X + (size_t)nb * F_DIM + f;
            const float4 a = *(const float4*)(src);
            const float4 b = *(const float4*)(src + 4);
            acc[0]+=a.x; acc[1]+=a.y; acc[2]+=a.z; acc[3]+=a.w;
            acc[4]+=b.x; acc[5]+=b.y; acc[6]+=b.z; acc[7]+=b.w;
        }
        #pragma unroll
        for (int i = 0; i < 8; ++i) acc[i] *= 0.1f;
    }
    #pragma unroll
    for (int i = 0; i < 8; ++i) dst[i] = (__bf16)acc[i];
}

// ---------------------------------------------------------------------------
// Kernel 2: Wt[n][k] = bf16(W[k][n]) via 32x32 LDS tile (both sides coalesced)
// ---------------------------------------------------------------------------
__global__ void __launch_bounds__(256)
transpose_w_kernel(const float* __restrict__ W, __bf16* __restrict__ Wt) {
    __shared__ float tile[32][33];
    const int tx = threadIdx.x;          // 0..31
    const int ty = threadIdx.y;          // 0..7
    const int n0 = blockIdx.x * 32;      // over O (1024)
    const int k0 = blockIdx.y * 32;      // over 2F (2048)

    #pragma unroll
    for (int j = 0; j < 4; ++j)
        tile[ty + j * 8][tx] = W[(size_t)(k0 + ty + j * 8) * O_DIM + (n0 + tx)];
    __syncthreads();
    #pragma unroll
    for (int j = 0; j < 4; ++j)
        Wt[(size_t)(n0 + ty + j * 8) * K2 + (k0 + tx)] =
            (__bf16)tile[tx][ty + j * 8];
}

// ---------------------------------------------------------------------------
// Kernel 3: out = combined[8192x2048] x Wt[1024x2048]^T + bias
// 128x128 block tile, K-step 32, double-buffered LDS filled by the CDNA5
// async copy engine (global_load_async_to_lds_b128, ASYNCcnt), so the copy of
// tile k+1 overlaps the 8 v_wmma_f32_16x16x32_bf16 of tile k. One barrier per
// K-step.
// ---------------------------------------------------------------------------
__global__ void __launch_bounds__(256)
sage_gemm_kernel(const __bf16* __restrict__ combined,
                 const __bf16* __restrict__ Wt,
                 const float* __restrict__ bias,
                 float* __restrict__ out) {
    __shared__ __bf16 As[2][128 * 32];
    __shared__ __bf16 Bs[2][128 * 32];

    const int tid    = threadIdx.x;
    const int lane   = tid & 31;
    const int wave   = tid >> 5;        // 0..7
    const int wm     = wave >> 2;       // 0..1 -> M offset wm*64
    const int wn     = wave & 3;        // 0..3 -> N offset wn*32
    const int l16    = lane & 15;
    const int khalf0 = (lane < 16) ? 0 : 8;   // first 16B chunk K-offset
    const int khalf1 = khalf0 + 16;           // second 16B chunk K-offset

    const int blockN = blockIdx.x * 128;
    const int blockM = blockIdx.y * 128;

    // Per-thread copy slots: 256 threads x 2 rounds x 16B cover one 128x32 tile.
    const int lin0 = tid;          // round 0
    const int lin1 = 256 + tid;    // round 1
    const int rowA0 = lin0 >> 2, colA0 = (lin0 & 3) * 8;
    const int rowA1 = lin1 >> 2, colA1 = (lin1 & 3) * 8;

    // Issue async global->LDS copies for K-tile at kk into buffer `buf`.
    auto issue_copy = [&](int buf, int kk) {
        const __bf16* gA0 = &combined[(size_t)(blockM + rowA0) * K2 + kk + colA0];
        const __bf16* gA1 = &combined[(size_t)(blockM + rowA1) * K2 + kk + colA1];
        const __bf16* gB0 = &Wt[(size_t)(blockN + rowA0) * K2 + kk + colA0];
        const __bf16* gB1 = &Wt[(size_t)(blockN + rowA1) * K2 + kk + colA1];
        // Low 32 bits of a generic pointer to __shared__ == wave-logical LDS
        // byte offset (flat aperture rule: the LDS address is the low 32 address bits).
        uint32_t lA0 = (uint32_t)(uintptr_t)&As[buf][rowA0 * 32 + colA0];
        uint32_t lA1 = (uint32_t)(uintptr_t)&As[buf][rowA1 * 32 + colA1];
        uint32_t lB0 = (uint32_t)(uintptr_t)&Bs[buf][rowA0 * 32 + colA0];
        uint32_t lB1 = (uint32_t)(uintptr_t)&Bs[buf][rowA1 * 32 + colA1];
        asm volatile("global_load_async_to_lds_b128 %0, %1, off"
                     :: "v"(lA0), "v"(gA0) : "memory");
        asm volatile("global_load_async_to_lds_b128 %0, %1, off"
                     :: "v"(lA1), "v"(gA1) : "memory");
        asm volatile("global_load_async_to_lds_b128 %0, %1, off"
                     :: "v"(lB0), "v"(gB0) : "memory");
        asm volatile("global_load_async_to_lds_b128 %0, %1, off"
                     :: "v"(lB1), "v"(gB1) : "memory");
    };

    v8f acc[4][2] = {};

    // Prime buffer 0.
    issue_copy(0, 0);
    asm volatile("s_wait_asynccnt 0x0" ::: "memory");
    __syncthreads();

    for (int kk = 0; kk < K2; kk += 32) {
        const int buf = (kk >> 5) & 1;
        if (kk + 32 < K2) issue_copy(buf ^ 1, kk + 32);   // overlaps the WMMAs

        // ---- LDS -> fragments (two ds_load_b128 per fragment) ----
        FragB16 afr[4], bfr[2];
        #pragma unroll
        for (int mi = 0; mi < 4; ++mi) {
            const int row = wm * 64 + mi * 16 + l16;
            afr[mi].h[0] = *(const v8bf*)&As[buf][row * 32 + khalf0];
            afr[mi].h[1] = *(const v8bf*)&As[buf][row * 32 + khalf1];
        }
        #pragma unroll
        for (int ni = 0; ni < 2; ++ni) {
            const int row = wn * 32 + ni * 16 + l16;
            bfr[ni].h[0] = *(const v8bf*)&Bs[buf][row * 32 + khalf0];
            bfr[ni].h[1] = *(const v8bf*)&Bs[buf][row * 32 + khalf1];
        }

        // ---- 8x v_wmma_f32_16x16x32_bf16 ----
        #pragma unroll
        for (int mi = 0; mi < 4; ++mi)
            #pragma unroll
            for (int ni = 0; ni < 2; ++ni)
                acc[mi][ni] = __builtin_amdgcn_wmma_f32_16x16x32_bf16(
                    false, afr[mi].v, false, bfr[ni].v,
                    (short)0, acc[mi][ni], false, false);

        // Next buffer's async copies must have landed before anyone reads it.
        asm volatile("s_wait_asynccnt 0x0" ::: "memory");
        __syncthreads();
    }

    // ---- epilogue: bias add + f32 store per C/D VGPR layout ----
    const int half = lane >> 4;   // 0: M rows 0-7, 1: M rows 8-15
    #pragma unroll
    for (int mi = 0; mi < 4; ++mi) {
        #pragma unroll
        for (int ni = 0; ni < 2; ++ni) {
            const int gm = blockM + wm * 64 + mi * 16 + half * 8;
            const int gn = blockN + wn * 32 + ni * 16 + l16;
            const float bv = bias[gn];
            #pragma unroll
            for (int r = 0; r < 8; ++r)
                out[(size_t)(gm + r) * O_DIM + gn] = acc[mi][ni][r] + bv;
        }
    }
}

// ---------------------------------------------------------------------------
extern "C" void kernel_launch(void* const* d_in, const int* in_sizes, int n_in,
                              void* d_out, int out_size, void* d_ws, size_t ws_size,
                              hipStream_t stream) {
    const float* X    = (const float*)d_in[0];
    // d_in[1] = A : dead code in the reference, never read.
    const float* W    = (const float*)d_in[2];
    const float* bias = (const float*)d_in[3];
    const int*   idx  = (const int*)d_in[4];
    float* out = (float*)d_out;

    char* ws = (char*)d_ws;
    __bf16* combined = (__bf16*)ws;                                   // 32 MB
    __bf16* Wt       = (__bf16*)(ws + (size_t)N_ROWS * K2 * 2);       //  4 MB

    // 1) gather-mean + concat -> bf16 combined [8192 x 2048]
    build_combined_kernel<<<N_ROWS, 256, 0, stream>>>(X, idx, combined);

    // 2) W [2048 x 1024] f32 -> Wt [1024 x 2048] bf16
    transpose_w_kernel<<<dim3(O_DIM / 32, K2 / 32), dim3(32, 8), 0, stream>>>(W, Wt);

    // 3) GEMM + bias -> out [8192 x 1024] f32
    sage_gemm_kernel<<<dim3(O_DIM / 128, N_ROWS / 128), 256, 0, stream>>>(
        combined, Wt, bias, out);
}
